// LocalMultiHeadChannelAttention_63883343560793
// MI455X (gfx1250) — matrix operand order repaired and checked by reading, not compiled, not run on hardware
//
#include <hip/hip_runtime.h>
#include <hip/hip_bf16.h>

// MI455X / gfx1250: wave32, WMMA 16x16x32 bf16 (f32 accumulate), TDM staging.
// B=32, C=256, R=64, H=16, D=256. NORM_C=0.1.
// Workspace layout (requires >= 128 MiB):
//   [0,            64 MiB) : V  bf16, layout (b,h,e,d) = z*65536 + e*256 + d
//   [64 MiB,      128 MiB) : Q  bf16, layout (b,h,c,e) = z*65536 + c*256 + e

typedef __attribute__((ext_vector_type(16))) __bf16 v16bf;
typedef __attribute__((ext_vector_type(8)))  float  v8f;
typedef __attribute__((ext_vector_type(4)))  unsigned int v4u;
typedef __attribute__((ext_vector_type(8)))  int v8i;
typedef __attribute__((ext_vector_type(4)))  int v4i;

#if __has_builtin(__builtin_amdgcn_tensor_load_to_lds) && \
    __has_builtin(__builtin_amdgcn_s_wait_tensorcnt)
#define HAVE_TDM 1
#else
#define HAVE_TDM 0
#endif

__device__ __forceinline__ __bf16 f2bf(float f) {
  unsigned u = __builtin_bit_cast(unsigned, f);
  u += 0x7fffu + ((u >> 16) & 1u);        // round-to-nearest-even
  unsigned short h = (unsigned short)(u >> 16);
  return __builtin_bit_cast(__bf16, h);
}

// A fragment (16xK=32 bf16), source tile is row-major [16][stride] in LDS.
// ISA 7.12.2: lanes 0-15 = rows, K 0..7 in v[0..7], K 16..23 in v[8..15];
// lanes 16-31 same rows, K 8..15 / 24..31.
__device__ __forceinline__ v16bf load_fragA(const __bf16* p0, int stride) {
  const int lane = threadIdx.x & 31;
  const int r    = lane & 15;
  const int koff = (lane >> 4) << 3;      // 0 or 8
  const __bf16* p = p0 + r * stride + koff;
  v16bf a;
#pragma unroll
  for (int i = 0; i < 8; ++i) a[i] = p[i];
#pragma unroll
  for (int i = 0; i < 8; ++i) a[i + 8] = p[i + 16];
  return a;
}

// B fragment (K=32 x 16 bf16), source is B^T row-major [16 n][stride k] in LDS.
// Lanes 0-15: column n, K 0..7 / 16..23; lanes 16-31: column n, K 8..15 / 24..31.
__device__ __forceinline__ v16bf load_fragB(const __bf16* p0, int stride) {
  const int lane = threadIdx.x & 31;
  const int n    = lane & 15;
  const int kb   = (lane >> 4) << 3;      // 0 or 8
  const __bf16* p = p0 + n * stride + kb;
  v16bf b;
#pragma unroll
  for (int i = 0; i < 8; ++i) b[i] = p[i];
#pragma unroll
  for (int i = 0; i < 8; ++i) b[i + 8] = p[i + 16];
  return b;
}

#define WMMA_BF16(accv, av, bv) \
  __builtin_amdgcn_wmma_f32_16x16x32_bf16(false, (av), false, (bv), (short)0, (accv), false, false)

#if HAVE_TDM
// Issue a TDM 2D tile load: tile_rows x tile_k bf16 elements from a row-major
// tensor (row length/stride = row_elts) into contiguous LDS at lds_off.
// D# packing per CDNA5 ISA 8.3/8.4 (group0 128b, group1 256b; groups 2/3 unused).
__device__ __forceinline__ void tdm_load_2d_bf16(unsigned lds_off, const __bf16* gsrc,
                                                 int tile_k, int tile_rows, int row_elts) {
  unsigned long long ga = (unsigned long long)(size_t)gsrc;
  v4u g0;
  g0[0] = 1u;                                           // count=1, user descriptor
  g0[1] = lds_off;                                      // lds_addr (bytes)
  g0[2] = (unsigned)ga;                                 // global_addr[31:0]
  g0[3] = (unsigned)((ga >> 32) & 0x01FFFFFFu) | (2u << 30);  // addr[56:32], type=2
  v8i g1;
  g1[0] = (int)(1u << 16);                              // data_size=1 (2 bytes)
  g1[1] = (int)(((unsigned)row_elts & 0xFFFFu) << 16);  // tensor_dim0[15:0]
  g1[2] = (int)((((unsigned)row_elts >> 16) & 0xFFFFu) |
                (((unsigned)tile_rows & 0xFFFFu) << 16)); // dim0 hi | tensor_dim1 lo
  g1[3] = (int)((((unsigned)tile_rows >> 16) & 0xFFFFu) |
                (((unsigned)tile_k & 0xFFFFu) << 16));    // dim1 hi | tile_dim0
  g1[4] = (int)((unsigned)tile_rows & 0xFFFFu);         // tile_dim1 (tile_dim2=0)
  g1[5] = row_elts;                                     // tensor_dim0_stride[31:0]
  g1[6] = 0;                                            // stride hi | dim1_stride lo
  g1[7] = 0;
  v4i gz = (v4i){0, 0, 0, 0};
#if defined(__clang_major__) && (__clang_major__ >= 23)
  v8i gz8 = (v8i){0, 0, 0, 0, 0, 0, 0, 0};
  __builtin_amdgcn_tensor_load_to_lds(g0, g1, gz, gz, gz8, 0);
#else
  __builtin_amdgcn_tensor_load_to_lds(g0, g1, gz, gz, 0);
#endif
}
#endif

// ---------------------------------------------------------------------------
// Kernel A: V = W_v @ x + b_v   (per batch: M=256 (o), N=4096 (s), K=256 (c))
// Output written bf16 in head-major layout: V[z=16b+h][e][d].
// ---------------------------------------------------------------------------
__global__ __launch_bounds__(256)
void wv_gemm_kernel(const float* __restrict__ x, const float* __restrict__ Wv,
                    const float* __restrict__ bv, __bf16* __restrict__ Vw) {
  __shared__ __bf16 ldsA[128 * 32];      // A tile row-major [m][k]
  __shared__ __bf16 ldsBT[128 * 32];     // B^T tile [n][k]
  const int b  = blockIdx.z;
  const int m0 = blockIdx.y * 128;       // o base
  const int n0 = blockIdx.x * 128;       // s base
  const int t  = threadIdx.x;
  const int w  = t >> 5, lane = t & 31;
  const int mw = (w & 3) * 32;           // 4x2 wave grid: 32x64 per wave
  const int nw = (w >> 2) * 64;

  const float* xb = x + (size_t)b * 256 * 4096;

  v8f z8 = {};
  v8f acc[2][4];
#pragma unroll
  for (int i = 0; i < 2; ++i)
#pragma unroll
    for (int j = 0; j < 4; ++j) acc[i][j] = z8;

  for (int kc = 0; kc < 8; ++kc) {
    const int k0 = kc * 32;
#pragma unroll
    for (int i = 0; i < 16; ++i) {       // stage A: Wv[m0+r][k0+c]
      int e = t + i * 256;
      int r = e >> 5, c = e & 31;
      ldsA[e] = f2bf(Wv[(m0 + r) * 256 + (k0 + c)]);
    }
#pragma unroll
    for (int i = 0; i < 16; ++i) {       // stage B^T: x[b][k0+kk][n0+ss]
      int e = t + i * 256;
      int kk = e >> 7, ss = e & 127;     // coalesced along s
      ldsBT[ss * 32 + kk] = f2bf(xb[(k0 + kk) * 4096 + (n0 + ss)]);
    }
    if (kc + 1 < 8) {                    // hint next chunk into cache
      __builtin_prefetch(&Wv[(m0 + (t >> 5) * 16) * 256 + (k0 + 32 + (t & 31))], 0, 3);
      __builtin_prefetch(&xb[(k0 + 32 + (t >> 7) * 16) * 4096 + (n0 + (t & 127))], 0, 3);
    }
    __syncthreads();
    v16bf af[2], bf[4];
#pragma unroll
    for (int i = 0; i < 2; ++i) af[i] = load_fragA(&ldsA[(mw + 16 * i) * 32], 32);
#pragma unroll
    for (int j = 0; j < 4; ++j) bf[j] = load_fragB(&ldsBT[(nw + 16 * j) * 32], 32);
#pragma unroll
    for (int i = 0; i < 2; ++i)
#pragma unroll
      for (int j = 0; j < 4; ++j) acc[i][j] = WMMA_BF16(acc[i][j], af[i], bf[j]);
    __syncthreads();
  }

  // Epilogue: +bias, remap (o,s) -> (h, e=(o%16)*16+s/256, d=s%256), store bf16.
  const int nl = lane & 15, hi = lane >> 4;
#pragma unroll
  for (int i = 0; i < 2; ++i) {
#pragma unroll
    for (int j = 0; j < 4; ++j) {
      int s = n0 + nw + 16 * j + nl;
#pragma unroll
      for (int r = 0; r < 8; ++r) {
        int o = m0 + mw + 16 * i + r + hi * 8;
        float v = acc[i][j][r] + bv[o];
        int h  = o >> 4;
        int e2 = ((o & 15) << 4) + (s >> 8);
        int d  = s & 255;
        Vw[(((b * 16 + h) * 256) + e2) * 256 + d] = f2bf(v);
      }
    }
  }
}

// ---------------------------------------------------------------------------
// Kernel B: Q = feat @ W_qk[h] + b_qk[h]  (per (b,h): M=N=K=256), bf16 out.
// ---------------------------------------------------------------------------
__global__ __launch_bounds__(256)
void qk_gemm_kernel(const float* __restrict__ x, const float* __restrict__ Wqk,
                    const float* __restrict__ bqk, __bf16* __restrict__ Qw) {
  __shared__ __bf16 ldsA[128 * 32];
  __shared__ __bf16 ldsBT[128 * 32];
  const int zz = blockIdx.z;             // b*16 + h
  const int h  = zz & 15;
  const int m0 = blockIdx.y * 128;       // c base
  const int n0 = blockIdx.x * 128;       // e base
  const int t  = threadIdx.x;
  const int w  = t >> 5, lane = t & 31;
  const int mw = (w & 3) * 32, nw = (w >> 2) * 64;

  const float* feat = x + (size_t)zz * 65536;   // (b,h) slab of x, row-major [c][d]
  const float* Wh   = Wqk + (size_t)h * 65536;  // [d][e] row-major

  v8f z8 = {};
  v8f acc[2][4];
#pragma unroll
  for (int i = 0; i < 2; ++i)
#pragma unroll
    for (int j = 0; j < 4; ++j) acc[i][j] = z8;

  for (int kc = 0; kc < 8; ++kc) {
    const int k0 = kc * 32;
#pragma unroll
    for (int i = 0; i < 16; ++i) {       // A: feat[m0+r][k0+c]
      int e = t + i * 256;
      int r = e >> 5, c = e & 31;
      ldsA[e] = f2bf(feat[(m0 + r) * 256 + (k0 + c)]);
    }
#pragma unroll
    for (int i = 0; i < 16; ++i) {       // B^T: Wh[k0+kk][n0+ee] -> [ee][kk]
      int e = t + i * 256;
      int kk = e >> 7, ee = e & 127;     // coalesced along e
      ldsBT[ee * 32 + kk] = f2bf(Wh[(k0 + kk) * 256 + (n0 + ee)]);
    }
    if (kc + 1 < 8) {
      __builtin_prefetch(&feat[((m0 + (t >> 5) * 16)) * 256 + (k0 + 32 + (t & 31))], 0, 3);
      __builtin_prefetch(&Wh[(k0 + 32 + (t >> 7) * 16) * 256 + (n0 + (t & 127))], 0, 3);
    }
    __syncthreads();
    v16bf af[2], bf[4];
#pragma unroll
    for (int i = 0; i < 2; ++i) af[i] = load_fragA(&ldsA[(mw + 16 * i) * 32], 32);
#pragma unroll
    for (int j = 0; j < 4; ++j) bf[j] = load_fragB(&ldsBT[(nw + 16 * j) * 32], 32);
#pragma unroll
    for (int i = 0; i < 2; ++i)
#pragma unroll
      for (int j = 0; j < 4; ++j) acc[i][j] = WMMA_BF16(acc[i][j], af[i], bf[j]);
    __syncthreads();
  }

  const int nl = lane & 15, hi = lane >> 4;
#pragma unroll
  for (int i = 0; i < 2; ++i) {
#pragma unroll
    for (int j = 0; j < 4; ++j) {
      int e = n0 + nw + 16 * j + nl;
      float bias = bqk[h * 256 + e];
#pragma unroll
      for (int r = 0; r < 8; ++r) {
        int c = m0 + mw + 16 * i + r + hi * 8;
        Qw[(size_t)zz * 65536 + c * 256 + e] = f2bf(acc[i][j][r] + bias);
      }
    }
  }
}

// ---------------------------------------------------------------------------
// Kernel C: per (b,h): S = Q Q^T; mean -> p -> denom; softmax; attn = W @ V;
// out = x + attn*(1+res_w).  512 threads = 16 waves; wave w owns S rows
// [16w, 16w+16) fully (16 tiles in registers) -> wave-local softmax.
// Phase-1 Q chunks are staged by the Tensor Data Mover (double-buffered:
// DMA of chunk kc+1 overlaps the WMMA burst on chunk kc).
// ---------------------------------------------------------------------------
__global__ __launch_bounds__(512)
void attn_kernel(const float* __restrict__ x, const __bf16* __restrict__ Qw,
                 const __bf16* __restrict__ Vw, const float* __restrict__ Wp,
                 const float* __restrict__ bp, const float* __restrict__ resw,
                 float* __restrict__ out) {
  __shared__ __bf16 ldsK[256 * 32];      // staging chunk; phase-5 V^T chunk
  __shared__ __bf16 ldsW[256 * 256];     // softmax weights (phase >=4); first
                                         // 16 KiB doubles as phase-1 buffer #2
  __shared__ float  rowsum[256];
  __shared__ float  invden[256];

  const int z = blockIdx.x;              // b*16 + h
  const int t = threadIdx.x;
  const int w = t >> 5, lane = t & 31;
  const __bf16* Qz = Qw + (size_t)z * 65536;
  const __bf16* Vz = Vw + (size_t)z * 65536;

  v8f z8 = {};
  v8f acc[16];
#pragma unroll
  for (int tc = 0; tc < 16; ++tc) acc[tc] = z8;

  // ---- Phase 1: S = Q Q^T (one chunk is both A and B^T: Q[r][k]) ----
#if HAVE_TDM
  {
    // NOTE: scalar locals + ternary selects only — LDS-derived pointers in
    // aggregate initializers become addrspacecast static initializers that
    // ld.lld cannot relocate.
    unsigned loffA = (unsigned)(size_t)&ldsK[0];
    unsigned loffB = (unsigned)(size_t)&ldsW[0];
    if (w == 0) {
      tdm_load_2d_bf16(loffA, Qz, 32, 256, 256);
      __builtin_amdgcn_s_wait_tensorcnt(0);
    }
    __syncthreads();
    int p = 0;
    for (int kc = 0; kc < 8; ++kc) {
      if (w == 0 && kc + 1 < 8)          // kick DMA for next chunk
        tdm_load_2d_bf16(p ? loffA : loffB, Qz + (kc + 1) * 32, 32, 256, 256);
      const __bf16* bufp = p ? &ldsW[0] : &ldsK[0];
      v16bf af = load_fragA(bufp + (16 * w) * 32, 32);
#pragma unroll
      for (int tc = 0; tc < 16; ++tc) {
        v16bf bf = load_fragB(bufp + (16 * tc) * 32, 32);
        acc[tc] = WMMA_BF16(acc[tc], af, bf);
      }
      if (w == 0) __builtin_amdgcn_s_wait_tensorcnt(0);
      __syncthreads();
      p ^= 1;
    }
  }
#else
  for (int kc = 0; kc < 8; ++kc) {
#pragma unroll
    for (int i = 0; i < 16; ++i) {
      int e = t + i * 512;               // 8192 bf16
      int r = e >> 5, c = e & 31;
      ldsK[e] = Qz[r * 256 + kc * 32 + c];
    }
    __syncthreads();
    v16bf af = load_fragA(&ldsK[(16 * w) * 32], 32);
#pragma unroll
    for (int tc = 0; tc < 16; ++tc) {
      v16bf bf = load_fragB(&ldsK[(16 * tc) * 32], 32);
      acc[tc] = WMMA_BF16(acc[tc], af, bf);
    }
    __syncthreads();
  }
#endif

  // ---- Phase 2: row sums (C-layout: VGPR j = rows 16w+j / 16w+j+8) ----
#pragma unroll
  for (int j = 0; j < 8; ++j) {
    float ps = 0.f;
#pragma unroll
    for (int tc = 0; tc < 16; ++tc) ps += acc[tc][j];
#pragma unroll
    for (int off = 1; off < 16; off <<= 1) ps += __shfl_xor(ps, off, 16);
    if (lane == 0)  rowsum[16 * w + j]     = ps;
    if (lane == 16) rowsum[16 * w + j + 8] = ps;
  }
  __syncthreads();

  // ---- Phase 3: p = sigmoid(mean @ Wp^T + bp); 1/denom = 2^(-8*(0.1+p)) ----
  if (t < 256) {
    float a = bp[t];
    const float* wr = Wp + t * 256;
    for (int c = 0; c < 256; ++c) a += (rowsum[c] * (1.0f / 256.0f)) * wr[c];
    float p = 1.0f / (1.0f + __expf(-a));
    invden[t] = exp2f(-8.0f * (0.1f + p));   // D = 256 = 2^8
  }
  __syncthreads();

  // ---- Phase 4: wave-local softmax over each owned row; weights -> LDS ----
#pragma unroll
  for (int j = 0; j < 8; ++j) {
    int row = 16 * w + j + ((lane >> 4) << 3);
    float sc = invden[row];
    float mx = -3.0e38f;
#pragma unroll
    for (int tc = 0; tc < 16; ++tc) {
      float v = acc[tc][j] * sc;
      acc[tc][j] = v;
      mx = fmaxf(mx, v);
    }
#pragma unroll
    for (int off = 1; off < 16; off <<= 1) mx = fmaxf(mx, __shfl_xor(mx, off, 16));
    float sm = 0.f;
#pragma unroll
    for (int tc = 0; tc < 16; ++tc) {
      float e = __expf(acc[tc][j] - mx);
      acc[tc][j] = e;
      sm += e;
    }
#pragma unroll
    for (int off = 1; off < 16; off <<= 1) sm += __shfl_xor(sm, off, 16);
    float inv = 1.0f / sm;
#pragma unroll
    for (int tc = 0; tc < 16; ++tc)
      ldsW[row * 256 + 16 * tc + (lane & 15)] = f2bf(acc[tc][j] * inv);
  }
  __syncthreads();

  // ---- Phase 5: attn = weights @ V  (stream V^T chunks through ldsK) ----
  v8f acc2[16];
#pragma unroll
  for (int td = 0; td < 16; ++td) acc2[td] = z8;

  for (int ec = 0; ec < 8; ++ec) {
    __syncthreads();
#pragma unroll
    for (int i = 0; i < 16; ++i) {
      int e = t + i * 512;
      int ek = e >> 8, d = e & 255;      // coalesced along d, transposed store
      ldsK[d * 32 + ek] = Vz[(32 * ec + ek) * 256 + d];
    }
    __syncthreads();
    v16bf af = load_fragA(&ldsW[(16 * w) * 256 + 32 * ec], 256);
#pragma unroll
    for (int td = 0; td < 16; ++td) {
      v16bf bf = load_fragB(&ldsK[(16 * td) * 32], 32);
      acc2[td] = WMMA_BF16(acc2[td], af, bf);
    }
  }

  // ---- Phase 6: out = x + attn*(1+res_w) ----
  const float rw = 1.0f + resw[0];
  const int nl = lane & 15, hi = lane >> 4;
  const float* xz = x + (size_t)z * 65536;
  float*       oz = out + (size_t)z * 65536;
#pragma unroll
  for (int td = 0; td < 16; ++td) {
    int d = 16 * td + nl;
#pragma unroll
    for (int j = 0; j < 8; ++j) {
      int c = 16 * w + j + hi * 8;
      int idx = c * 256 + d;
      oz[idx] = xz[idx] + acc2[td][j] * rw;
    }
  }
}

// ---------------------------------------------------------------------------
extern "C" void kernel_launch(void* const* d_in, const int* in_sizes, int n_in,
                              void* d_out, int out_size, void* d_ws, size_t ws_size,
                              hipStream_t stream) {
  const float* x    = (const float*)d_in[0];
  const float* Wqk  = (const float*)d_in[1];
  const float* bqk  = (const float*)d_in[2];
  const float* Wp   = (const float*)d_in[3];
  const float* bp   = (const float*)d_in[4];
  const float* Wv   = (const float*)d_in[5];
  const float* bv   = (const float*)d_in[6];
  const float* resw = (const float*)d_in[7];
  float* out = (float*)d_out;

  __bf16* Vw = (__bf16*)d_ws;                          // 32*16*256*256 bf16 = 64 MiB
  __bf16* Qw = Vw + (size_t)32 * 16 * 256 * 256;       // + 64 MiB

  // V = Wv @ x + bv : per batch M=256, N=4096 -> tiles 2 x 32
  wv_gemm_kernel<<<dim3(32, 2, 32), 256, 0, stream>>>(x, Wv, bv, Vw);
  // Q = feat @ Wqk[h] + bqk[h] : per (b,h) M=N=256 -> tiles 2 x 2
  qk_gemm_kernel<<<dim3(2, 2, 512), 256, 0, stream>>>(x, Wqk, bqk, Qw);
  // fused channel attention, one workgroup per (b,h)
  attn_kernel<<<dim3(512), 512, 0, stream>>>(x, Qw, Vw, Wp, bp, resw, out);
}
